// QuantizedLinear_377957122271
// MI455X (gfx1250) — matrix operand order repaired and checked by reading, not compiled, not run on hardware
//
#include <hip/hip_runtime.h>

typedef __attribute__((ext_vector_type(16))) _Float16 v16h;
typedef __attribute__((ext_vector_type(8)))  _Float16 v8h;
typedef __attribute__((ext_vector_type(8)))  float    v8f;
typedef __attribute__((ext_vector_type(4)))  float    v4f;
typedef __attribute__((ext_vector_type(4)))  int      v4i;

constexpr int QM = 16384;   // batch*seq = 8*2048
constexpr int QN = 4096;    // out_features
constexpr int QK = 4096;    // in_features

constexpr int TILE_M = 128;
constexpr int TILE_N = 128;
constexpr int TILE_K = 64;  // two K=32 WMMA phases per tile
constexpr int LDT    = 72;  // f16 stride per LDS row (144B: 16B-aligned, conflict-free b128 reads)

__global__ __launch_bounds__(256)
void qlinear_wmma_f16(const float* __restrict__ input,
                      const int*   __restrict__ wq,
                      const float* __restrict__ wscale_p,
                      const int*   __restrict__ wzp_p,
                      const int*   __restrict__ bq,
                      const float* __restrict__ bscale_p,
                      const int*   __restrict__ bzp_p,
                      float* __restrict__ out)
{
    // Double-buffered LDS tiles: 2 * (128*72*2B) per matrix = ~73.7 KB total.
    __shared__ __align__(16) _Float16 lA[2][TILE_M * LDT];
    __shared__ __align__(16) _Float16 lB[2][TILE_N * LDT];

    const int tid  = threadIdx.x;
    const int lane = tid & 31;
    const int wave = tid >> 5;
    const int l    = lane & 15;
    const int half = lane >> 4;

    const int nBase = blockIdx.x * TILE_N;
    const int mBase = blockIdx.y * TILE_M;

    const int waveM = wave & 3;    // 0..3 -> 32 rows each
    const int waveN = wave >> 2;   // 0..1 -> 64 cols each

    const int   izp    = wzp_p[0];
    const float wscale = wscale_p[0];

    // Global staging: 256 threads cover a 128x64 tile, 32 elems/thread/matrix.
    const int seg  = tid & 7;      // group of 8 along K (8 groups = 64)
    const int row0 = tid >> 3;     // 0..31 ; rows row0 + 32*i, i=0..3

    v4f aReg[4][2];
    v4i bReg[4][2];

    auto loadTile = [&](int k0) {
        #pragma unroll
        for (int i = 0; i < 4; ++i) {
            const int ra = mBase + row0 + 32 * i;
            const float* pa = input + (size_t)ra * QK + k0 + seg * 8;
            aReg[i][0] = *(const v4f*)pa;
            aReg[i][1] = *(const v4f*)(pa + 4);
            const int rb = nBase + row0 + 32 * i;
            const int* pb = wq + (size_t)rb * QK + k0 + seg * 8;
            bReg[i][0] = *(const v4i*)pb;
            bReg[i][1] = *(const v4i*)(pb + 4);
        }
    };

    auto stashTile = [&](int buf) {
        #pragma unroll
        for (int i = 0; i < 4; ++i) {
            v8h ha, hb;
            #pragma unroll
            for (int j = 0; j < 4; ++j) {
                ha[j]     = (_Float16)aReg[i][0][j];
                ha[j + 4] = (_Float16)aReg[i][1][j];
                // (q - zp) is an integer in [-130,126]: exact in f16.
                hb[j]     = (_Float16)(bReg[i][0][j] - izp);
                hb[j + 4] = (_Float16)(bReg[i][1][j] - izp);
            }
            *(v8h*)&lA[buf][(row0 + 32 * i) * LDT + seg * 8] = ha;
            *(v8h*)&lB[buf][(row0 + 32 * i) * LDT + seg * 8] = hb;
        }
    };

    v8f acc[2][4];
    #pragma unroll
    for (int mi = 0; mi < 2; ++mi)
        #pragma unroll
        for (int ni = 0; ni < 4; ++ni)
            acc[mi][ni] = (v8f){0.f, 0.f, 0.f, 0.f, 0.f, 0.f, 0.f, 0.f};

    constexpr int KTILES = QK / TILE_K;   // 64

    loadTile(0);
    stashTile(0);
    __syncthreads();

    for (int kt = 0; kt < KTILES; ++kt) {
        const int buf = kt & 1;

        // Issue next tile's global loads early; they retire under the WMMAs.
        if (kt + 1 < KTILES) loadTile((kt + 1) * TILE_K);
        if (kt + 2 < KTILES) {
            const int k2 = (kt + 2) * TILE_K;
            __builtin_prefetch(input + (size_t)(mBase + row0) * QK + k2 + seg * 8, 0, 3);
            __builtin_prefetch(wq    + (size_t)(nBase + row0) * QK + k2 + seg * 8, 0, 3);
        }

        // Two K=32 WMMA phases over the 64-wide tile.
        #pragma unroll
        for (int kp = 0; kp < 2; ++kp) {
            const int kb = kp * 32;  // f16 offset into the row

            // A fragments (ISA 16-bit A layout: lane-half K groups {0-7,16-23}/{8-15,24-31})
            v16h afrag[2];
            #pragma unroll
            for (int mi = 0; mi < 2; ++mi) {
                const int rowA = waveM * 32 + mi * 16 + l;
                const v8h a0 = *(const v8h*)&lA[buf][rowA * LDT + kb + half * 8];
                const v8h a1 = *(const v8h*)&lA[buf][rowA * LDT + kb + 16 + half * 8];
                #pragma unroll
                for (int j = 0; j < 8; ++j) { afrag[mi][j] = a0[j]; afrag[mi][j + 8] = a1[j]; }
            }
            // B fragments (same K order as A: lane-half K groups {0-7,16-23}/{8-15,24-31})
            v16h bfrag[4];
            #pragma unroll
            for (int ni = 0; ni < 4; ++ni) {
                const int rowB = waveN * 64 + ni * 16 + l;
                const v8h b0 = *(const v8h*)&lB[buf][rowB * LDT + kb + half * 8];
                const v8h b1 = *(const v8h*)&lB[buf][rowB * LDT + kb + 16 + half * 8];
                #pragma unroll
                for (int j = 0; j < 8; ++j) { bfrag[ni][j] = b0[j]; bfrag[ni][j + 8] = b1[j]; }
            }

            #pragma unroll
            for (int mi = 0; mi < 2; ++mi)
                #pragma unroll
                for (int ni = 0; ni < 4; ++ni)
                    acc[mi][ni] = __builtin_amdgcn_wmma_f32_16x16x32_f16(
                        /*neg_a=*/false, afrag[mi],
                        /*neg_b=*/false, bfrag[ni],
                        /*c_mod=*/(short)0, acc[mi][ni],
                        /*reuse_a=*/false, /*reuse_b=*/false);
        }

        // Convert + stash next tile into the other buffer (safe: all reads of
        // buf^1 were retired by the barrier at the end of iteration kt-1).
        if (kt + 1 < KTILES) stashTile(buf ^ 1);
        __syncthreads();
    }

    // Epilogue: out = wscale*acc + dequant(bias)
    const float bscale = bscale_p[0];
    const float bzp    = (float)bzp_p[0];
    float bias[4];
    #pragma unroll
    for (int ni = 0; ni < 4; ++ni) {
        const int n = nBase + waveN * 64 + ni * 16 + l;
        bias[ni] = ((float)bq[n] - bzp) * bscale;
    }

    #pragma unroll
    for (int mi = 0; mi < 2; ++mi) {
        #pragma unroll
        for (int ni = 0; ni < 4; ++ni) {
            const int n = nBase + waveN * 64 + ni * 16 + l;
            #pragma unroll
            for (int j = 0; j < 8; ++j) {
                const int m = mBase + waveM * 32 + mi * 16 + half * 8 + j;
                out[(size_t)m * QN + n] = wscale * acc[mi][ni][j] + bias[ni];
            }
        }
    }
}

extern "C" void kernel_launch(void* const* d_in, const int* in_sizes, int n_in,
                              void* d_out, int out_size, void* d_ws, size_t ws_size,
                              hipStream_t stream) {
    const float* input  = (const float*)d_in[0];
    const int*   wq     = (const int*)  d_in[1];
    const float* wscale = (const float*)d_in[2];
    const int*   wzp    = (const int*)  d_in[3];
    const int*   bq     = (const int*)  d_in[4];
    const float* bscale = (const float*)d_in[5];
    const int*   bzp    = (const int*)  d_in[6];
    float* out = (float*)d_out;

    dim3 grid(QN / TILE_N, QM / TILE_M);   // (32, 128)
    qlinear_wmma_f16<<<grid, 256, 0, stream>>>(input, wq, wscale, wzp, bq, bscale, bzp, out);
}
